// yolowrapper_63350767616292
// MI455X (gfx1250) — 8-bit matrix operand order repaired and checked by reading, compiled, not run on hardware
//
#include <hip/hip_runtime.h>
#include <stdint.h>

#define Bn 16
#define Nn 8400
#define Cn 80
#define Gn 20
#define Kn 1000
#define KP 1008
#define Mn 16384

typedef int v8i __attribute__((ext_vector_type(8)));

__global__ __launch_bounds__(256) void k_conf(const float* __restrict__ scores,
                                              unsigned long long* __restrict__ keys,
                                              int* __restrict__ clsout) {
  int id = blockIdx.x * blockDim.x + threadIdx.x;
  if (id >= Bn * Mn) return;
  int b = id >> 14;
  int n = id & (Mn - 1);
  if (n >= Nn) { keys[id] = 0ull; return; }

  const float4* sp = (const float4*)(scores + ((size_t)b * Nn + n) * Cn);
  __builtin_prefetch((const void*)(sp + 20), 0, 1);
  float best = -1.0f; int bc = 0;
#pragma unroll
  for (int q = 0; q < 20; ++q) {
    float4 v = sp[q];
    if (v.x > best) { best = v.x; bc = 4 * q + 0; }
    if (v.y > best) { best = v.y; bc = 4 * q + 1; }
    if (v.z > best) { best = v.z; bc = 4 * q + 2; }
    if (v.w > best) { best = v.w; bc = 4 * q + 3; }
  }
  float conf = (best >= 0.25f) ? best : 0.0f;
  unsigned long long key = ((unsigned long long)__float_as_uint(conf) << 32)
                         | (unsigned long long)(0xFFFFFFFFu - (unsigned)n);
  keys[id] = key;
  clsout[(size_t)b * Nn + n] = bc;
}

__global__ __launch_bounds__(1024) void k_sort(unsigned long long* __restrict__ keys) {
  unsigned long long* g = keys + (size_t)blockIdx.x * Mn;
  int t = threadIdx.x;
  for (int k = 2; k <= Mn; k <<= 1) {
    for (int j = k >> 1; j > 0; j >>= 1) {
      for (int e = 0; e < Mn / 1024; ++e) {
        int i = e * 1024 + t;
        int l = i ^ j;
        if (l > i) {
          unsigned long long a = g[i], c = g[l];
          bool sw = ((i & k) == 0) ? (a < c) : (a > c);
          if (sw) { g[i] = c; g[l] = a; }
        }
      }
      __syncthreads();
    }
  }
}

__global__ __launch_bounds__(1024) void k_nms_match(
    const float* __restrict__ boxes, const float* __restrict__ gt_boxes,
    const int* __restrict__ gt_labels, const unsigned long long* __restrict__ keys,
    const int* __restrict__ clsarr, float* __restrict__ out) {
  __shared__ float x1s[KP], y1s[KP], x2s[KP], y2s[KP], vals[KP];
  __shared__ int clss[KP];
  __shared__ unsigned char keeps[KP];
  __shared__ unsigned int maskg[KP];
  __shared__ float gbx[Gn * 4];
  __shared__ int gls[Gn];
  __shared__ float r_iou[Gn], r_conf[Gn], r_mat[Gn];

  int b = blockIdx.x;
  int t = threadIdx.x;

  if (t < KP) {
    maskg[t] = 0u;
    if (t < Kn) {
      unsigned long long key = keys[(size_t)b * Mn + t];
      float v = __uint_as_float((unsigned)(key >> 32));
      unsigned idx = 0xFFFFFFFFu - (unsigned)(key & 0xFFFFFFFFu);
      const float* bp = boxes + ((size_t)b * Nn + idx) * 4;
      float cx = bp[0], cy = bp[1], w = bp[2], h = bp[3];
      x1s[t] = cx - w * 0.5f; y1s[t] = cy - h * 0.5f;
      x2s[t] = cx + w * 0.5f; y2s[t] = cy + h * 0.5f;
      clss[t] = clsarr[(size_t)b * Nn + idx];
      vals[t] = v;
      keeps[t] = (v > 0.0f) ? 1 : 0;
    } else {
      x1s[t] = y1s[t] = x2s[t] = y2s[t] = 0.0f;
      vals[t] = 0.0f; clss[t] = -1; keeps[t] = 0;
    }
  }
  if (t < Gn) {
    gbx[t * 4 + 0] = gt_boxes[((size_t)b * Gn + t) * 4 + 0];
    gbx[t * 4 + 1] = gt_boxes[((size_t)b * Gn + t) * 4 + 1];
    gbx[t * 4 + 2] = gt_boxes[((size_t)b * Gn + t) * 4 + 2];
    gbx[t * 4 + 3] = gt_boxes[((size_t)b * Gn + t) * 4 + 3];
    gls[t] = gt_labels[(size_t)b * Gn + t];
  }
  __syncthreads();

  for (int i = 0; i < Kn; ++i) {
    if (keeps[i] && t < Kn && t > i) {
      float offi = (float)clss[i] * 7680.0f;
      float offj = (float)clss[t] * 7680.0f;
      float ax1 = x1s[i] + offi, ay1 = y1s[i] + offi;
      float ax2 = x2s[i] + offi, ay2 = y2s[i] + offi;
      float bx1 = x1s[t] + offj, by1 = y1s[t] + offj;
      float bx2 = x2s[t] + offj, by2 = y2s[t] + offj;
      float xx1 = fmaxf(ax1, bx1), yy1 = fmaxf(ay1, by1);
      float xx2 = fminf(ax2, bx2), yy2 = fminf(ay2, by2);
      float inter = fmaxf(xx2 - xx1, 0.0f) * fmaxf(yy2 - yy1, 0.0f);
      float areaA = (ax2 - ax1) * (ay2 - ay1);
      float areaB = (bx2 - bx1) * (by2 - by1);
      float iou = inter / (areaA + areaB - inter + 1e-7f);
      if (iou > 0.45f) keeps[t] = 0;
    }
    __syncthreads();
  }

  int lane = t & 31;
  int wid = t >> 5;
  int hf = lane >> 4;
  int ln = lane & 15;
  for (int tp = wid; tp < 126; tp += 32) {
    int mt = tp >> 1;
    int nt = tp & 1;
    int ai = mt * 16 + ln;
    int ci = (ai < KP && keeps[ai]) ? clss[ai] : -1;
    int gg = nt * 16 + ln;
    int lb = (gg < Gn) ? gls[gg] : -1;
    v8i a0, a1, b0, b1;
#pragma unroll
    for (int v = 0; v < 8; ++v) {
      int ka = ((v >> 1) << 4) + (hf << 3) + ((v & 1) << 2);
      int d0 = ci - ka;
      int d1 = ci - (ka + 64);
      a0[v] = (d0 >= 0 && d0 < 4) ? (int)(1u << (8 * d0)) : 0;
      a1[v] = (d1 >= 0 && d1 < 4) ? (int)(1u << (8 * d1)) : 0;
      int kb = ((v >> 1) << 4) + (hf << 3) + ((v & 1) << 2);
      int e0 = lb - kb;
      int e1 = lb - (kb + 64);
      b0[v] = (e0 >= 0 && e0 < 4) ? (int)(1u << (8 * e0)) : 0;
      b1[v] = (e1 >= 0 && e1 < 4) ? (int)(1u << (8 * e1)) : 0;
    }
    v8i acc = {0, 0, 0, 0, 0, 0, 0, 0};
    acc = __builtin_amdgcn_wmma_i32_16x16x64_iu8(false, a0, false, b0, acc, false, false);
    acc = __builtin_amdgcn_wmma_i32_16x16x64_iu8(false, a1, false, b1, acc, false, false);
#pragma unroll
    for (int r = 0; r < 8; ++r) {
      int mi = mt * 16 + r + (hf << 3);
      int gd = nt * 16 + ln;
      if (acc[r] != 0 && mi < KP && gd < Gn) atomicOr(&maskg[mi], 1u << gd);
    }
  }
  __syncthreads();

  if (wid < Gn) {
    int g = wid;
    float gx1 = gbx[g * 4 + 0], gy1 = gbx[g * 4 + 1];
    float gx2 = gbx[g * 4 + 2], gy2 = gbx[g * 4 + 3];
    float gArea = (gx2 - gx1) * (gy2 - gy1);
    float bestv = -2.0f; int besti = 0x7FFFFFFF;
    for (int i = lane; i < Kn; i += 32) {
      float iou_m = -1.0f;
      if ((maskg[i] >> g) & 1u) {
        float xx1 = fmaxf(x1s[i], gx1), yy1 = fmaxf(y1s[i], gy1);
        float xx2 = fminf(x2s[i], gx2), yy2 = fminf(y2s[i], gy2);
        float inter = fmaxf(xx2 - xx1, 0.0f) * fmaxf(yy2 - yy1, 0.0f);
        float areaA = (x2s[i] - x1s[i]) * (y2s[i] - y1s[i]);
        iou_m = inter / (areaA + gArea - inter + 1e-7f);
      }
      if (iou_m > bestv) { bestv = iou_m; besti = i; }
    }
    for (int off = 16; off > 0; off >>= 1) {
      float ov = __shfl_xor(bestv, off, 32);
      int oi = __shfl_xor(besti, off, 32);
      if (ov > bestv || (ov == bestv && oi < besti)) { bestv = ov; besti = oi; }
    }
    if (lane == 0) {
      float bconf = vals[besti];
      r_iou[g] = bestv;
      r_conf[g] = bconf;
      r_mat[g] = (bestv > 0.6f && bconf > 0.5f) ? 1.0f : 0.0f;
    }
  }
  __syncthreads();

  if (t < Gn) {
    out[b * Gn + t] = r_iou[t];
    out[Bn * Gn + b * Gn + t] = r_conf[t];
    out[2 * Bn * Gn + b * Gn + t] = r_mat[t];
  }
  if (t == 0) {
    float cnt = 0.0f, s0 = 0.0f, s1 = 0.0f;
    for (int g = 0; g < Gn; ++g) {
      cnt += r_mat[g];
      s0 += r_iou[g] * r_mat[g];
      s1 += r_conf[g] * r_mat[g];
    }
    float denom = fmaxf(cnt, 1.0f);
    float* st = out + 3 * Bn * Gn + b * 3;
    st[0] = s0 / denom;
    st[1] = s1 / denom;
    st[2] = cnt / (float)Gn;
  }
}

extern "C" void kernel_launch(void* const* d_in, const int* in_sizes, int n_in,
                              void* d_out, int out_size, void* d_ws, size_t ws_size,
                              hipStream_t stream) {
  const float* boxes  = (const float*)d_in[0];
  const float* scores = (const float*)d_in[1];
  const float* gtb    = (const float*)d_in[2];
  const int*   gtl    = (const int*)d_in[3];

  unsigned long long* keys = (unsigned long long*)d_ws;
  int* clsws = (int*)((char*)d_ws + (size_t)Bn * Mn * sizeof(unsigned long long));

  int total = Bn * Mn;
  k_conf<<<(total + 255) / 256, 256, 0, stream>>>(scores, keys, clsws);
  k_sort<<<Bn, 1024, 0, stream>>>(keys);
  k_nms_match<<<Bn, 1024, 0, stream>>>(boxes, gtb, gtl, keys, clsws, (float*)d_out);
}
